// VanillaSamplingCanonical_47141561041223
// MI455X (gfx1250) — matrix operand order repaired and checked by reading, not compiled, not run on hardware
//
#include <hip/hip_runtime.h>
#include <hip/hip_bf16.h>

// ---------------------------------------------------------------------------
// NeRF forward (coarse + fine) for MI455X / gfx1250.
//  - wave32 WMMA f16->f32 (v_wmma_f32_16x16x32_f16), M=64 per block, 8 waves
//  - weights pre-swizzled into B-fragment lane order (32B contiguous per lane)
//  - weight K-slabs staged global->LDS with the Tensor Data Mover
//    (tensor_load_to_lds + s_wait_tensorcnt), double buffered
// ---------------------------------------------------------------------------

typedef __attribute__((ext_vector_type(16))) _Float16 v16h;
typedef __attribute__((ext_vector_type(8)))  _Float16 v8h;
typedef __attribute__((ext_vector_type(8)))  float    v8f;
typedef __attribute__((ext_vector_type(4)))  unsigned v4u;
typedef __attribute__((ext_vector_type(8)))  int      v8i_t;
typedef __attribute__((ext_vector_type(4)))  int      v4i_t;

#define N_RAYS        1024
#define N_SAMPLES     64
#define N_IMPORTANCE  128
#define N_TOTAL       192
#define NEARP         2.0f
#define FARP          6.0f

// Swizzled-weight half-offsets (order: W0..W7, W_feat, W_rgb1)
#define H_W0    0
#define H_W1    24576
#define H_W2    90112
#define H_W3    155648
#define H_W4    221184
#define H_W5    286720
#define H_W6    376832
#define H_W7    442368
#define H_WFEAT 507904
#define H_WRGB1 573440
#define H_TOTAL 610304

// Workspace byte offsets
#define OFF_W    0u
#define OFF_ZC   1220608u                // 1024*64 f32
#define OFF_ZF   (OFF_ZC + 262144u)      // 1024*192 f32
#define OFF_RAWC (OFF_ZF + 786432u)      // 65536*4 f32
#define OFF_RAWF (OFF_RAWC + 1048576u)   // 196608*4 f32

// Dynamic-LDS arena byte offsets (per block, M = 64 points)
#define L_X     0u        // 64 x 352 f16 : [h0(84) | h4(256) | pad]
#define L_A     45056u    // 64 x 256 f16
#define L_B     77824u    // 64 x 256 f16
#define L_D     110592u   // 64 x 32  f16 : dir_emb(27) | pad
#define L_WS0   114688u   // 16KB weight slab 0
#define L_WS1   131072u   // 16KB weight slab 1
#define L_SIG   147456u   // 64 f32
#define LDS_BYTES 147712u

__device__ __forceinline__ float rand01(unsigned x) {
  x ^= x >> 17; x *= 0xED5AD4BBu;
  x ^= x >> 11; x *= 0xAC4C1B51u;
  x ^= x >> 15; x *= 0x31848BABu;
  x ^= x >> 14;
  return (float)(x >> 8) * (1.0f / 16777216.0f);
}

// LDS byte offset of a generic pointer known to point into LDS.
__device__ __forceinline__ unsigned lds_off(const void* p) {
  return (unsigned)(size_t)(__attribute__((address_space(3))) const void*)p;
}

// ---------------------------------------------------------------------------
// TDM: 1-D global->LDS copy of nbytes (multiple of 8), built per ISA D# spec:
// group0: count=1, lds_addr, global_addr, type=2
// group1: data_size=3 (8B), tensor_dim0=tile_dim0=n8, tensor_dim1=tile_dim1=1,
//         tensor_dim0_stride=n8
// ---------------------------------------------------------------------------
__device__ __forceinline__ void tdm_load_1d(unsigned lds_byte,
                                            const void* gptr,
                                            unsigned nbytes) {
  unsigned long long ga = (unsigned long long)(size_t)gptr;
  unsigned n8 = nbytes >> 3;
  v4u g0;
  g0[0] = 1u;                                    // count=1, is_restore=0
  g0[1] = lds_byte;                              // lds_addr (bits 63:32)
  g0[2] = (unsigned)(ga & 0xffffffffu);          // global_addr[31:0]
  g0[3] = (unsigned)((ga >> 32) & 0x01ffffffu)   // global_addr[56:32]
        | (2u << 30);                            // type=2 ("image")
  v8i_t g1;
  g1[0] = (int)(3u << 16);                       // wg_mask=0, data_size=3 (8B)
  g1[1] = (int)((n8 & 0xffffu) << 16);           // tensor_dim0[15:0] @ bits 63:48
  g1[2] = (int)(((n8 >> 16) & 0xffffu)           // tensor_dim0[31:16]
        | (1u << 16));                           // tensor_dim1[15:0] = 1
  g1[3] = (int)((n8 & 0xffffu) << 16);           // tile_dim0 @ bits 127:112
  g1[4] = 1;                                     // tile_dim1=1, tile_dim2=0
  g1[5] = (int)n8;                               // tensor_dim0_stride[31:0]
  g1[6] = 0;                                     // stride[47:32]=0, dim1_stride lo=0
  g1[7] = 0;
  v4i_t z4 = {};
#if defined(__clang_major__) && (__clang_major__ >= 23)
  v8i_t z8 = {};
  __builtin_amdgcn_tensor_load_to_lds(g0, g1, z4, z4, z8, 0);
#else
  __builtin_amdgcn_tensor_load_to_lds(g0, g1, z4, z4, 0);
#endif
}

// ---------------------------------------------------------------------------
// Weight prep: f32 [K][N] row-major -> f16 fragments in B-operand lane order.
// Fragment (kt,ct) holds 32(K)x16(N); lane l supplies 16 halves:
//   element h -> B[kt*32 + 16*(l>>4) + h][ct*16 + (l&15)]
// ---------------------------------------------------------------------------
struct WSrc { const float* p[10]; };

__global__ void prep_weights(WSrc src, _Float16* __restrict__ dst) {
  constexpr int Kreal[10] = {84,256,256,256,256,340,256,256,256,283};
  constexpr int Ncol[10]  = {256,256,256,256,256,256,256,256,256,128};
  constexpr int Hoff[10]  = {H_W0,H_W1,H_W2,H_W3,H_W4,H_W5,H_W6,H_W7,H_WFEAT,H_WRGB1};
  int idx = blockIdx.x * blockDim.x + threadIdx.x;
  if (idx >= H_TOTAL) return;
  int m = 0;
  while (m + 1 < 10 && idx >= Hoff[m + 1]) m++;
  int local = idx - Hoff[m];
  int frag  = local >> 9;
  int lane  = (local >> 4) & 31;
  int h     = local & 15;
  int Nt    = Ncol[m] >> 4;
  int kt    = frag / Nt;
  int ct    = frag - kt * Nt;
  int col   = ct * 16 + (lane & 15);
  int k     = kt * 32 + 16 * (lane >> 4) + h;
  float v = (k < Kreal[m]) ? src.p[m][(size_t)k * Ncol[m] + col] : 0.0f;
  dst[idx] = (_Float16)v;
}

// ---------------------------------------------------------------------------
// Stratified coarse depths with hash jitter.
// ---------------------------------------------------------------------------
__global__ void gen_z_coarse(float* __restrict__ zc) {
  int i = blockIdx.x * blockDim.x + threadIdx.x;
  if (i >= N_RAYS * N_SAMPLES) return;
  int s = i & (N_SAMPLES - 1);
  const float step = (FARP - NEARP) / (float)(N_SAMPLES - 1);
  float zs = NEARP + step * (float)s;
  float lower = (s == 0) ? zs : zs - 0.5f * step;
  float upper = (s == N_SAMPLES - 1) ? zs : zs + 0.5f * step;
  float t = rand01(0x9E3779B9u * (unsigned)i + 0x85EBCA6Bu);
  zc[i] = lower + (upper - lower) * t;
}

// ---------------------------------------------------------------------------
// WMMA MLP. Block = 256 threads (8 waves), 64 points per block.
// ---------------------------------------------------------------------------
struct MlpArgs {
  const float* rays;
  const int*   indices;
  const int*   max_index;
  const float* z;          // [N_RAYS][S] flat
  const _Float16* wz;      // swizzled weights
  const float* bias[10];   // b0..b7, b_feat, b_rgb1
  const float* w_sigma;
  const float* b_sigma;
  const float* w_rgb2;
  const float* b_rgb2;
  float*       raw;        // [npts][4]
  int          S;
};

// A fragment: 16x32 f16 tile from row-major LDS, row stride ld halves.
__device__ __forceinline__ v16h load_a_frag(const _Float16* __restrict__ act,
                                            int ld, int rowbase, int kbase,
                                            int lane) {
  int row = rowbase + (lane & 15);
  int g   = lane >> 4;
  const _Float16* p = act + row * ld + kbase + 16 * g;
  v8h lo = *(const v8h*)p;
  v8h hi = *(const v8h*)(p + 8);
  v16h r;
#pragma unroll
  for (int i = 0; i < 8; ++i) { r[i] = lo[i]; r[i + 8] = hi[i]; }
  return r;
}

// One GEMM layer: out[64][N] = act(in @ W + b).
// K may be split across two input buffers at k-tile 'ksw' (feat|dir concat).
// Weight K-slabs (Nt*512 halves each) are TDM-staged into LDS, double buffered.
__device__ __forceinline__ void gemm_layer(
    const _Float16* __restrict__ wzL, const float* __restrict__ bias,
    const _Float16* in0, int ld0, const _Float16* in1, int ld1, int ksw,
    _Float16* out, int ldo, int ocol,
    _Float16* slab0, _Float16* slab1,
    int Kt, int Nt, int ni, bool do_relu,
    int lane, int wave) {
  const int slab_halves = Nt * 512;
  v8f acc[2][4];
#pragma unroll
  for (int i = 0; i < 2; ++i)
#pragma unroll
    for (int rt = 0; rt < 4; ++rt) acc[i][rt] = (v8f){};

  if (wave == 0)
    tdm_load_1d(lds_off(slab0), wzL, (unsigned)slab_halves * 2u);

#pragma unroll 1
  for (int kt = 0; kt < Kt; ++kt) {
    if (wave == 0) __builtin_amdgcn_s_wait_tensorcnt(0);
    __syncthreads();  // slab(kt) visible to all waves; prior compute finished
    _Float16* slab  = (kt & 1) ? slab1 : slab0;
    _Float16* nslab = (kt & 1) ? slab0 : slab1;
    if (wave == 0 && kt + 1 < Kt)
      tdm_load_1d(lds_off(nslab), wzL + (size_t)(kt + 1) * slab_halves,
                  (unsigned)slab_halves * 2u);

    const _Float16* src = (kt < ksw) ? in0 : in1;
    int ld = (kt < ksw) ? ld0 : ld1;
    int kb = (kt < ksw) ? kt * 32 : (kt - ksw) * 32;

    v16h a[4];
#pragma unroll
    for (int rt = 0; rt < 4; ++rt)
      a[rt] = load_a_frag(src, ld, rt * 16, kb, lane);

#pragma unroll
    for (int i = 0; i < 2; ++i) {
      if (i < ni) {
        int ct = wave + 8 * i;
        v16h b = *(const v16h*)(slab + ct * 512 + lane * 16);
#pragma unroll
        for (int rt = 0; rt < 4; ++rt)
          acc[i][rt] = __builtin_amdgcn_wmma_f32_16x16x32_f16(
              false, a[rt], false, b, (short)0, acc[i][rt], false, false);
      }
    }
    __syncthreads();  // compute(kt) done before slab overwrite next iteration
  }

  int g = lane >> 4;
#pragma unroll
  for (int i = 0; i < 2; ++i) {
    if (i < ni) {
      int ct = wave + 8 * i;
      int n  = ocol + ct * 16 + (lane & 15);
      float bv = bias[ct * 16 + (lane & 15)];
#pragma unroll
      for (int rt = 0; rt < 4; ++rt) {
#pragma unroll
        for (int r = 0; r < 8; ++r) {
          float v = acc[i][rt][r] + bv;
          if (do_relu) v = fmaxf(v, 0.0f);
          out[(rt * 16 + 8 * g + r) * ldo + n] = (_Float16)v;
        }
      }
    }
  }
}

__global__ __launch_bounds__(256) void nerf_mlp(MlpArgs a) {
  extern __shared__ __align__(32) char smem[];
  _Float16* bufX = (_Float16*)(smem + L_X);
  _Float16* bufA = (_Float16*)(smem + L_A);
  _Float16* bufB = (_Float16*)(smem + L_B);
  _Float16* bufD = (_Float16*)(smem + L_D);
  _Float16* ws0  = (_Float16*)(smem + L_WS0);
  _Float16* ws1  = (_Float16*)(smem + L_WS1);
  float*    sig  = (float*)(smem + L_SIG);

  const int tid  = threadIdx.x;
  const int lane = tid & 31;
  const int wave = tid >> 5;
  const int p0   = blockIdx.x * 64;

  for (int i = tid; i < 64 * 352; i += 256) bufX[i] = (_Float16)0.0f;
  for (int i = tid; i < 64 * 32;  i += 256) bufD[i] = (_Float16)0.0f;
  __syncthreads();

  if (tid < 64) {
    int p = p0 + tid;
    int r = p / a.S;
    float o0 = a.rays[r * 6 + 0], o1 = a.rays[r * 6 + 1], o2 = a.rays[r * 6 + 2];
    float d0 = a.rays[r * 6 + 3], d1 = a.rays[r * 6 + 4], d2 = a.rays[r * 6 + 5];
    float z  = a.z[p];
    float x[3] = {o0 + d0 * z, o1 + d1 * z, o2 + d2 * z};
    float dv[3] = {d0, d1, d2};
    _Float16* row = bufX + tid * 352;
    row[0] = (_Float16)x[0]; row[1] = (_Float16)x[1]; row[2] = (_Float16)x[2];
#pragma unroll 1
    for (int l = 0; l < 10; ++l) {
      float f = (float)(1 << l);
#pragma unroll
      for (int j = 0; j < 3; ++j) {
        row[3 + 6 * l + j]     = (_Float16)__sinf(x[j] * f);
        row[3 + 6 * l + 3 + j] = (_Float16)__cosf(x[j] * f);
      }
    }
    float ts0 = (float)a.indices[0] / (float)a.max_index[0];
    row[63] = (_Float16)ts0;
#pragma unroll 1
    for (int l = 0; l < 10; ++l) {
      float f = (float)(1 << l);
      row[64 + 2 * l] = (_Float16)__sinf(ts0 * f);
      row[65 + 2 * l] = (_Float16)__cosf(ts0 * f);
    }
    _Float16* drow = bufD + tid * 32;
    drow[0] = (_Float16)dv[0]; drow[1] = (_Float16)dv[1]; drow[2] = (_Float16)dv[2];
#pragma unroll 1
    for (int l = 0; l < 4; ++l) {
      float f = (float)(1 << l);
#pragma unroll
      for (int j = 0; j < 3; ++j) {
        drow[3 + 6 * l + j]     = (_Float16)__sinf(dv[j] * f);
        drow[3 + 6 * l + 3 + j] = (_Float16)__cosf(dv[j] * f);
      }
    }
  }
  __syncthreads();

  const _Float16* W = a.wz;
  // L0: h0(96) -> bufA
  gemm_layer(W + H_W0, a.bias[0], bufX, 352, bufX, 352, 3, bufA, 256, 0,  ws0, ws1, 3, 16, 2, true, lane, wave);
  __syncthreads();
  gemm_layer(W + H_W1, a.bias[1], bufA, 256, bufA, 256, 8, bufB, 256, 0,  ws0, ws1, 8, 16, 2, true, lane, wave);
  __syncthreads();
  gemm_layer(W + H_W2, a.bias[2], bufB, 256, bufB, 256, 8, bufA, 256, 0,  ws0, ws1, 8, 16, 2, true, lane, wave);
  __syncthreads();
  gemm_layer(W + H_W3, a.bias[3], bufA, 256, bufA, 256, 8, bufB, 256, 0,  ws0, ws1, 8, 16, 2, true, lane, wave);
  __syncthreads();
  // L4: write into bufX cols 84..339 (skip concat: bufX = [h0 | h4])
  gemm_layer(W + H_W4, a.bias[4], bufB, 256, bufB, 256, 8, bufX, 352, 84, ws0, ws1, 8, 16, 2, true, lane, wave);
  __syncthreads();
  // L5: K = 352 (340 real) from bufX
  gemm_layer(W + H_W5, a.bias[5], bufX, 352, bufX, 352, 11, bufA, 256, 0, ws0, ws1, 11, 16, 2, true, lane, wave);
  __syncthreads();
  gemm_layer(W + H_W6, a.bias[6], bufA, 256, bufA, 256, 8, bufB, 256, 0,  ws0, ws1, 8, 16, 2, true, lane, wave);
  __syncthreads();
  gemm_layer(W + H_W7, a.bias[7], bufB, 256, bufB, 256, 8, bufA, 256, 0,  ws0, ws1, 8, 16, 2, true, lane, wave);
  __syncthreads();

  // sigma head (N=1): VALU dot product per point
  if (tid < 64) {
    float sv = a.b_sigma[0];
#pragma unroll 1
    for (int k = 0; k < 256; ++k) sv += (float)bufA[tid * 256 + k] * a.w_sigma[k];
    sig[tid] = sv;
  }
  __syncthreads();

  // feat: linear (no relu) bufA -> bufB
  gemm_layer(W + H_WFEAT, a.bias[8], bufA, 256, bufA, 256, 8, bufB, 256, 0, ws0, ws1, 8, 16, 2, false, lane, wave);
  __syncthreads();
  // rgb1: K = 288 = feat(256, bufB) | dir(32, bufD), N = 128 -> bufA
  gemm_layer(W + H_WRGB1, a.bias[9], bufB, 256, bufD, 32, 8, bufA, 256, 0, ws0, ws1, 9, 8, 1, true, lane, wave);
  __syncthreads();

  // rgb2 head (N=3) + pack raw output
  if (tid < 64) {
    int p = p0 + tid;
#pragma unroll
    for (int j = 0; j < 3; ++j) {
      float v = a.b_rgb2[j];
#pragma unroll 1
      for (int k = 0; k < 128; ++k) v += (float)bufA[tid * 256 + k] * a.w_rgb2[k * 3 + j];
      a.raw[(size_t)p * 4 + j] = v;
    }
    a.raw[(size_t)p * 4 + 3] = sig[tid];
  }
}

// ---------------------------------------------------------------------------
// Coarse volume rendering -> PDF importance sampling -> sorted z_fine.
// One block (64 threads) per ray.
// ---------------------------------------------------------------------------
__global__ __launch_bounds__(64) void render_coarse(
    const float* __restrict__ rays, const float* __restrict__ zc,
    const float* __restrict__ rawc, float* __restrict__ zf) {
  __shared__ float z[N_SAMPLES];
  __shared__ float w[N_SAMPLES];
  __shared__ float bins[N_SAMPLES - 1];
  __shared__ float cdf[N_SAMPLES - 1];
  __shared__ float znew[N_IMPORTANCE];
  __shared__ float zall[N_TOTAL];

  int r = blockIdx.x;
  int t = threadIdx.x;
  if (t < N_SAMPLES) z[t] = zc[r * N_SAMPLES + t];
  __syncthreads();

  float d0 = rays[r * 6 + 3], d1 = rays[r * 6 + 4], d2 = rays[r * 6 + 5];
  float nrm = sqrtf(d0 * d0 + d1 * d1 + d2 * d2);

  if (t < N_SAMPLES) {
    float sg = fmaxf(rawc[(size_t)(r * N_SAMPLES + t) * 4 + 3], 0.0f);
    float delta = ((t < N_SAMPLES - 1) ? (z[t + 1] - z[t]) : 1e10f) * nrm;
    w[t] = 1.0f - expf(-sg * delta);
  }
  if (t < N_SAMPLES - 1) bins[t] = 0.5f * (z[t] + z[t + 1]);
  __syncthreads();

  if (t == 0) {
    float T = 1.0f;
    for (int s = 0; s < N_SAMPLES; ++s) {
      float al = w[s];
      w[s] = T * al;
      T *= (1.0f - al + 1e-10f);
    }
    float sum = 0.0f;
    for (int s = 1; s <= 62; ++s) sum += w[s] + 1e-5f;
    cdf[0] = 0.0f;
    float c = 0.0f;
    for (int i = 0; i < 62; ++i) {
      c += (w[i + 1] + 1e-5f) / sum;
      cdf[i + 1] = c;
    }
  }
  __syncthreads();

  for (int ii = t; ii < N_IMPORTANCE; ii += 64) {
    float u = rand01(0xB5297A4Du * (unsigned)(r * N_IMPORTANCE + ii) + 0x68E31DA4u);
    int lo = 0, hi = 63;
    while (lo < hi) {
      int mid = (lo + hi) >> 1;
      if (u >= cdf[mid]) lo = mid + 1; else hi = mid;
    }
    int below = min(max(lo - 1, 0), 62);
    int above = min(lo, 62);
    float c0 = cdf[below], c1 = cdf[above];
    float b0 = bins[below], b1 = bins[above];
    float den = (c1 - c0 < 1e-5f) ? 1.0f : (c1 - c0);
    float tt = (u - c0) / den;
    znew[ii] = b0 + tt * (b1 - b0);
  }
  __syncthreads();

  for (int ii = t; ii < N_TOTAL; ii += 64)
    zall[ii] = (ii < N_SAMPLES) ? z[ii] : znew[ii - N_SAMPLES];
  __syncthreads();

  for (int pass = 0; pass < N_TOTAL; ++pass) {
    int par = pass & 1;
    for (int i = par + 2 * t; i < N_TOTAL - 1; i += 128) {
      float a0 = zall[i], a1 = zall[i + 1];
      if (a0 > a1) { zall[i] = a1; zall[i + 1] = a0; }
    }
    __syncthreads();
  }
  for (int ii = t; ii < N_TOTAL; ii += 64) zf[r * N_TOTAL + ii] = zall[ii];
}

// ---------------------------------------------------------------------------
// Final volume rendering: one thread per ray.
// out[0..3071] = rgb_map, out[3072..4095] = depth_map.
// ---------------------------------------------------------------------------
__global__ void render_fine(const float* __restrict__ rays,
                            const float* __restrict__ zf,
                            const float* __restrict__ rawf,
                            float* __restrict__ out) {
  int r = blockIdx.x * blockDim.x + threadIdx.x;
  if (r >= N_RAYS) return;
  float d0 = rays[r * 6 + 3], d1 = rays[r * 6 + 4], d2 = rays[r * 6 + 5];
  float nrm = sqrtf(d0 * d0 + d1 * d1 + d2 * d2);
  float T = 1.0f, a0 = 0.0f, a1 = 0.0f, a2 = 0.0f, dep = 0.0f;
  for (int s = 0; s < N_TOTAL; ++s) {
    float zv = zf[r * N_TOTAL + s];
    float dist = ((s < N_TOTAL - 1) ? (zf[r * N_TOTAL + s + 1] - zv) : 1e10f) * nrm;
    const float* raw = rawf + (size_t)(r * N_TOTAL + s) * 4;
    float sg = fmaxf(raw[3], 0.0f);
    float al = 1.0f - expf(-sg * dist);
    float wt = T * al;
    T *= (1.0f - al + 1e-10f);
    a0 += wt / (1.0f + expf(-raw[0]));
    a1 += wt / (1.0f + expf(-raw[1]));
    a2 += wt / (1.0f + expf(-raw[2]));
    dep += wt * zv;
  }
  out[r * 3 + 0] = a0;
  out[r * 3 + 1] = a1;
  out[r * 3 + 2] = a2;
  out[3 * N_RAYS + r] = dep;
}

// ---------------------------------------------------------------------------
// Launch. Input order (setup_inputs dict order, params in insertion order):
//  0 rays, 1 indices, 2 max_index,
//  3+2i = W{i}, 4+2i = b{i} (i=0..7),
//  19 W_sigma, 20 b_sigma, 21 W_feat, 22 b_feat,
//  23 W_rgb1, 24 b_rgb1, 25 W_rgb2, 26 b_rgb2
// ---------------------------------------------------------------------------
extern "C" void kernel_launch(void* const* d_in, const int* in_sizes, int n_in,
                              void* d_out, int out_size, void* d_ws, size_t ws_size,
                              hipStream_t stream) {
  (void)in_sizes; (void)n_in; (void)out_size; (void)ws_size;
  const float* rays      = (const float*)d_in[0];
  const int*   indices   = (const int*)d_in[1];
  const int*   max_index = (const int*)d_in[2];

  WSrc src;
  for (int i = 0; i < 8; ++i) src.p[i] = (const float*)d_in[3 + 2 * i];
  src.p[8] = (const float*)d_in[21];  // W_feat
  src.p[9] = (const float*)d_in[23];  // W_rgb1

  char* ws = (char*)d_ws;
  _Float16* wz = (_Float16*)(ws + OFF_W);
  float* zc    = (float*)(ws + OFF_ZC);
  float* zfv   = (float*)(ws + OFF_ZF);
  float* rawc  = (float*)(ws + OFF_RAWC);
  float* rawf  = (float*)(ws + OFF_RAWF);

  (void)hipFuncSetAttribute((const void*)nerf_mlp,
                            hipFuncAttributeMaxDynamicSharedMemorySize,
                            (int)LDS_BYTES);

  prep_weights<<<(H_TOTAL + 255) / 256, 256, 0, stream>>>(src, wz);
  gen_z_coarse<<<(N_RAYS * N_SAMPLES + 255) / 256, 256, 0, stream>>>(zc);

  MlpArgs ac;
  ac.rays = rays; ac.indices = indices; ac.max_index = max_index;
  ac.z = zc; ac.wz = wz;
  for (int i = 0; i < 8; ++i) ac.bias[i] = (const float*)d_in[4 + 2 * i];
  ac.bias[8] = (const float*)d_in[22];  // b_feat
  ac.bias[9] = (const float*)d_in[24];  // b_rgb1
  ac.w_sigma = (const float*)d_in[19];
  ac.b_sigma = (const float*)d_in[20];
  ac.w_rgb2  = (const float*)d_in[25];
  ac.b_rgb2  = (const float*)d_in[26];
  ac.raw = rawc; ac.S = N_SAMPLES;
  nerf_mlp<<<(N_RAYS * N_SAMPLES) / 64, 256, LDS_BYTES, stream>>>(ac);

  render_coarse<<<N_RAYS, 64, 0, stream>>>(rays, zc, rawc, zfv);

  MlpArgs af = ac;
  af.z = zfv; af.raw = rawf; af.S = N_TOTAL;
  nerf_mlp<<<(N_RAYS * N_TOTAL) / 64, 256, LDS_BYTES, stream>>>(af);

  render_fine<<<(N_RAYS + 63) / 64, 64, 0, stream>>>(rays, zfv, rawf, (float*)d_out);
}
